// BasketBSDESolver_54348516163992
// MI455X (gfx1250) — compile-verified, matrix operand order checked by reading, not run on hardware
//
#include <hip/hip_runtime.h>
#include <math.h>

// ---------------- problem constants ----------------
#define DIMA   100      // assets
#define MSTEP  50       // time steps
#define HDIM   128      // hidden
#define BATCH  8192
#define NPAD4  112      // DIM padded to 16-multiple for WMMA N tiles (7 tiles)
#define SROW   104      // padded S row stride (floats)
#define ROWS   64       // rows per block (4 waves x 16)
#define S0INV  0.01f
#define RDT    0.001f   // R*DT = 0.05 * 0.02
#define SIG    0.2f
#define LNEPS  1e-5f

typedef __attribute__((ext_vector_type(16))) _Float16 v16h;
typedef __attribute__((ext_vector_type(8)))  _Float16 v8h;
typedef __attribute__((ext_vector_type(8)))  float    v8f;

union V16U { v16h v; v8h h[2]; };

// Load one 16x32 f16 WMMA fragment (A-row or B-col striped) from LDS.
// base points at the start of this lane's row/col (stride 128 halves);
// kb = kt*32 + half*8. Two 16-byte LDS loads per fragment.
__device__ __forceinline__ v16h load_frag(const _Float16* base, int kb) {
  V16U u;
  u.h[0] = *(const v8h*)(base + kb);
  u.h[1] = *(const v8h*)(base + kb + 16);
  return u.v;
}

// butterfly sum over each 16-lane half of the wave32
__device__ __forceinline__ float red16(float x) {
#pragma unroll
  for (int off = 1; off < 16; off <<= 1) x += __shfl_xor(x, off, 16);
  return x;
}

// One hidden layer: out = GELU(LayerNorm(in @ W + b) * g + be), all via WMMA.
// Wave computes its own 16-row strip; no cross-wave communication.
__device__ __forceinline__ void mlp_layer(const _Float16* in, const _Float16* Wt,
                                          const float* bb, const float* gg,
                                          const float* ee, _Float16* out,
                                          int rbase, int lane15, int half) {
  v8f acc[8];
#pragma unroll
  for (int nt = 0; nt < 8; ++nt) acc[nt] = {};

  const _Float16* arow = in + (rbase + lane15) * HDIM;
#pragma unroll
  for (int kt = 0; kt < 4; ++kt) {
    const int kb = kt * 32 + half * 8;
    v16h a = load_frag(arow, kb);
#pragma unroll
    for (int nt = 0; nt < 8; ++nt) {
      v16h b = load_frag(Wt + (nt * 16 + lane15) * HDIM, kb);
      acc[nt] = __builtin_amdgcn_wmma_f32_16x16x32_f16(
          false, a, false, b, (short)0, acc[nt], false, false);
    }
  }

  float bn[8], gn[8], en[8];
#pragma unroll
  for (int nt = 0; nt < 8; ++nt) {
    const int n = nt * 16 + lane15;
    bn[nt] = bb[n]; gn[nt] = gg[n]; en[nt] = ee[n];
  }

#pragma unroll
  for (int j = 0; j < 8; ++j) {
    float p = 0.f;
#pragma unroll
    for (int nt = 0; nt < 8; ++nt) { acc[nt][j] += bn[nt]; p += acc[nt][j]; }
    p = red16(p);
    const float mu = p * (1.f / 128.f);
    float q = 0.f;
#pragma unroll
    for (int nt = 0; nt < 8; ++nt) { float d = acc[nt][j] - mu; q += d * d; }
    q = red16(q);
    const float rstd = rsqrtf(q * (1.f / 128.f) + LNEPS);

    const int row = rbase + j + half * 8;
    _Float16* orow = out + row * HDIM + lane15;
#pragma unroll
    for (int nt = 0; nt < 8; ++nt) {
      float x = (acc[nt][j] - mu) * rstd * gn[nt] + en[nt];
      float ge = 0.5f * x * (1.f + erff(x * 0.70710678118654752f));
      orow[nt * 16] = (_Float16)ge;
    }
  }
}

extern "C" __global__ void __launch_bounds__(128)
bsde_fused_kernel(const float* __restrict__ dw, const float* __restrict__ tg,
                  const float* __restrict__ W1, const float* __restrict__ b1,
                  const float* __restrict__ g1, const float* __restrict__ be1,
                  const float* __restrict__ W2, const float* __restrict__ b2,
                  const float* __restrict__ g2, const float* __restrict__ be2,
                  const float* __restrict__ W3, const float* __restrict__ b3,
                  const float* __restrict__ g3, const float* __restrict__ be3,
                  const float* __restrict__ W4, const float* __restrict__ b4,
                  const float* __restrict__ Y0,
                  float* __restrict__ outY, float* __restrict__ outS) {
  extern __shared__ __align__(16) char smem[];
  _Float16* Wt1 = (_Float16*)smem;                    // [128][128]
  _Float16* Wt2 = Wt1 + HDIM * HDIM;                  // [128][128]
  _Float16* Wt3 = Wt2 + HDIM * HDIM;                  // [128][128]
  _Float16* Wt4 = Wt3 + HDIM * HDIM;                  // [112][128]
  float* prm = (float*)(Wt4 + NPAD4 * HDIM);          // 9*128 + 112 floats
  float* pb1 = prm;        float* pg1 = prm + 128;  float* pe1 = prm + 256;
  float* pb2 = prm + 384;  float* pg2 = prm + 512;  float* pe2 = prm + 640;
  float* pb3 = prm + 768;  float* pg3 = prm + 896;  float* pe3 = prm + 1024;
  float* pb4 = prm + 1152;                            // [112]
  float* Sl  = prm + 1264;                            // [64][104]
  float* Yl  = Sl + ROWS * SROW;                      // [64]
  _Float16* buf0 = (_Float16*)(Yl + ROWS);            // [64][128]
  _Float16* buf1 = buf0 + ROWS * HDIM;                // [64][128]

  const int tid = threadIdx.x;
  const int lane = tid & 31, lane15 = lane & 15, half = lane >> 4;
  const int rbase = (tid >> 5) * 16;                  // wave's first row
  const long blockBase = (long)blockIdx.x * ROWS;

  // ---- stage weights (fp32 -> fp16, transposed [n][k], zero-padded) ----
  for (int i = tid; i < HDIM * HDIM; i += 128) {
    const int n = i >> 7, k = i & 127;
    Wt1[i] = (_Float16)(k < DIMA + 1 ? W1[k * HDIM + n] : 0.f);
    Wt2[i] = (_Float16)W2[k * HDIM + n];
    Wt3[i] = (_Float16)W3[k * HDIM + n];
  }
  for (int i = tid; i < NPAD4 * HDIM; i += 128) {
    const int n = i >> 7, k = i & 127;
    Wt4[i] = (_Float16)(n < DIMA ? W4[k * DIMA + n] : 0.f);
  }
  pb1[tid] = b1[tid]; pg1[tid] = g1[tid]; pe1[tid] = be1[tid];
  pb2[tid] = b2[tid]; pg2[tid] = g2[tid]; pe2[tid] = be2[tid];
  pb3[tid] = b3[tid]; pg3[tid] = g3[tid]; pe3[tid] = be3[tid];
  if (tid < NPAD4) pb4[tid] = (tid < DIMA) ? b4[tid] : 0.f;

  // ---- init per-wave state ----
  const float y0 = Y0[0];
#pragma unroll 4
  for (int r = 0; r < 16; ++r) {
    for (int d = lane; d < DIMA; d += 32) Sl[(rbase + r) * SROW + d] = 100.f;
    if (lane == 0) Yl[rbase + r] = y0;
  }
  __syncthreads();

  // ---- 50-step recurrence, fully on-chip ----
  for (int m = 0; m < MSTEP; ++m) {
    // build network input x = [S/S0, t] (fp16) for this wave's rows
    for (int i = lane; i < 16 * HDIM; i += 32) {
      const int rl = i >> 7, k = i & 127;
      const int row = rbase + rl;
      float v;
      if (k < DIMA)      v = Sl[row * SROW + k] * S0INV;
      else if (k == DIMA) v = tg[(blockBase + row) * MSTEP + m];
      else               v = 0.f;
      buf0[row * HDIM + k] = (_Float16)v;
    }

    mlp_layer(buf0, Wt1, pb1, pg1, pe1, buf1, rbase, lane15, half);
    mlp_layer(buf1, Wt2, pb2, pg2, pe2, buf0, rbase, lane15, half);
    mlp_layer(buf0, Wt3, pb3, pg3, pe3, buf1, rbase, lane15, half);

    // ---- head: zeta = sigmoid(h @ W4 + b4), then SDE update ----
    {
      v8f acc[7];
#pragma unroll
      for (int nt = 0; nt < 7; ++nt) acc[nt] = {};
      const _Float16* arow = buf1 + (rbase + lane15) * HDIM;
#pragma unroll
      for (int kt = 0; kt < 4; ++kt) {
        const int kb = kt * 32 + half * 8;
        v16h a = load_frag(arow, kb);
#pragma unroll
        for (int nt = 0; nt < 7; ++nt) {
          v16h b = load_frag(Wt4 + (nt * 16 + lane15) * HDIM, kb);
          acc[nt] = __builtin_amdgcn_wmma_f32_16x16x32_f16(
              false, a, false, b, (short)0, acc[nt], false, false);
        }
      }
      float bn[7];
#pragma unroll
      for (int nt = 0; nt < 7; ++nt) bn[nt] = pb4[nt * 16 + lane15];

#pragma unroll
      for (int j = 0; j < 8; ++j) {
        const int row = rbase + j + half * 8;
        const long bidx = blockBase + row;
        const float* dwr = dw + ((long)bidx * MSTEP + m) * DIMA;
        float gain = 0.f, asset = 0.f;
#pragma unroll
        for (int nt = 0; nt < 7; ++nt) {
          const int d = nt * 16 + lane15;
          if (d < DIMA) {
            const float z  = 1.f / (1.f + expf(-(acc[nt][j] + bn[nt])));
            const float s  = Sl[row * SROW + d];
            const float ds = s * (RDT + SIG * dwr[d]);
            gain  += z * ds;
            asset += z * s;
            Sl[row * SROW + d] = s + ds;
          }
        }
        gain  = red16(gain);
        asset = red16(asset);
        if (lane15 == 0) {
          const float y = Yl[row];
          Yl[row] = y + RDT * (y - asset) + gain;
        }
      }
    }
  }

  // ---- write outputs: d_out = [Y (8192) | S (8192*100)] ----
  __syncthreads();
  for (int i = tid; i < ROWS; i += 128) outY[blockBase + i] = Yl[i];
  for (int i = tid; i < ROWS * DIMA; i += 128) {
    const int r = i / DIMA, d = i - r * DIMA;
    outS[(blockBase + r) * (long)DIMA + d] = Sl[r * SROW + d];
  }
}

extern "C" void kernel_launch(void* const* d_in, const int* in_sizes, int n_in,
                              void* d_out, int out_size, void* d_ws, size_t ws_size,
                              hipStream_t stream) {
  (void)in_sizes; (void)n_in; (void)out_size; (void)d_ws; (void)ws_size;
  const float* dw  = (const float*)d_in[0];
  const float* tg  = (const float*)d_in[1];
  const float* W1  = (const float*)d_in[2];
  const float* b1  = (const float*)d_in[3];
  const float* g1  = (const float*)d_in[4];
  const float* be1 = (const float*)d_in[5];
  const float* W2  = (const float*)d_in[6];
  const float* b2  = (const float*)d_in[7];
  const float* g2  = (const float*)d_in[8];
  const float* be2 = (const float*)d_in[9];
  const float* W3  = (const float*)d_in[10];
  const float* b3  = (const float*)d_in[11];
  const float* g3  = (const float*)d_in[12];
  const float* be3 = (const float*)d_in[13];
  const float* W4  = (const float*)d_in[14];
  const float* b4  = (const float*)d_in[15];
  const float* Y0  = (const float*)d_in[16];

  float* outY = (float*)d_out;
  float* outS = outY + BATCH;

  // LDS: 3*32KB + 28KB weights + 5KB params + 26KB S + Y + 2*16KB act = 191,680 B
  const size_t smem =
      (size_t)(3 * HDIM * HDIM + NPAD4 * HDIM) * sizeof(_Float16) +
      (size_t)(9 * 128 + NPAD4) * sizeof(float) +
      (size_t)(ROWS * SROW + ROWS) * sizeof(float) +
      (size_t)(2 * ROWS * HDIM) * sizeof(_Float16);

  dim3 grid(BATCH / ROWS);   // 128 blocks
  dim3 block(128);           // 4 waves, each owns 16 rows
  bsde_fused_kernel<<<grid, block, smem, stream>>>(
      dw, tg, W1, b1, g1, be1, W2, b2, g2, be2, W3, b3, g3, be3, W4, b4, Y0,
      outY, outS);
}
